// TreeBLSTM_16071767622152
// MI455X (gfx1250) — matrix operand order repaired and checked by reading, not compiled, not run on hardware
//
#include <hip/hip_runtime.h>
#include <hip/hip_bf16.h>
#include <stdint.h>

#define IN_DIM 512
#define HID    512
#define OUTD   256
#define BRANCH 4
#define DEPTH  8
#define KTOT   1024   // IN+HID (hidden GEMMs) == 2*HID (output GEMM)

typedef __attribute__((ext_vector_type(16))) __bf16 v16bf;
typedef __attribute__((ext_vector_type(8)))  float  v8f;
typedef unsigned short u16;
typedef unsigned int   u32;

// ---------------------------------------------------------------- helpers
__device__ __forceinline__ u16 f2bf(float f) {
  u32 u = __float_as_uint(f);
  u32 r = u + 0x7fffu + ((u >> 16) & 1u);   // round-to-nearest-even
  return (u16)(r >> 16);
}
__device__ __forceinline__ float bf2f(u16 h) {
  return __uint_as_float(((u32)h) << 16);
}

// gfx1250 async LDS-direct load: 16B per lane, memory -> LDS, ASYNCcnt.
__device__ __forceinline__ void async_load_b128(u32 ldsByteOff, const void* g) {
  asm volatile("global_load_async_to_lds_b128 %0, %1, off"
               :: "v"(ldsByteOff), "v"((unsigned long long)(uintptr_t)g)
               : "memory");
}
__device__ __forceinline__ void wait_async8() {
  asm volatile("s_wait_asynccnt 0x8" ::: "memory");
}
__device__ __forceinline__ void wait_async0() {
  asm volatile("s_wait_asynccnt 0x0" ::: "memory");
}

// ---------------------------------------------------------------- converts
__global__ void f32_to_bf16_kernel(const float* __restrict__ src,
                                   u16* __restrict__ dst, int n) {
  int i = blockIdx.x * blockDim.x + threadIdx.x;
  if (i < n) dst[i] = f2bf(src[i]);
}

// mean of 4 child rows; operates on packed dwords (2 bf16 each).
// hcatChild: (4*Nl x 1024) bf16 ; hm: (Nl x 1024) bf16 ; nDw = Nl*512
__global__ void child_mean_kernel(const u32* __restrict__ hcatChild,
                                  u32* __restrict__ hm, int nDw) {
  int i = blockIdx.x * blockDim.x + threadIdx.x;
  if (i >= nDw) return;
  int m = i >> 9;          // row (1024 bf16 = 512 dwords per row)
  int c = i & 511;         // dword column
  const u32* base = hcatChild + (size_t)(4 * m) * 512 + c;
  float lo = 0.f, hi = 0.f;
#pragma unroll
  for (int j = 0; j < 4; ++j) {
    u32 w = base[(size_t)j * 512];
    lo += bf2f((u16)(w & 0xffffu));
    hi += bf2f((u16)(w >> 16));
  }
  hm[i] = (u32)f2bf(lo * 0.25f) | ((u32)f2bf(hi * 0.25f) << 16);
}

// ---------------------------------------------------------------- WMMA GEMM
// C[m,n] = act( sum_k A(m,k) * W(n,k) + bias[n] )
// A split in two K-halves: A0 covers k in [0,512), A1 covers [512,1024).
// A1 == nullptr => second half is all zeros (leaf level) -> skipped.
// W: (N x 1024) bf16 row-major (n-major) == B[k,n] operand of WMMA.
#define BM 128
#define BN 128
#define KC 64
#define LDT 72   // padded LDS row stride in bf16 elements (144B, 16B aligned)
#define TILE_ELEMS ((BM + BN) * LDT)       // one double-buffer slot
#define TILE_BYTES (TILE_ELEMS * 2)

union Frag { u32 u[8]; v16bf v; };

template <bool HIDDEN>
__global__ void __launch_bounds__(256)
wmma_gemm_kernel(const u16* __restrict__ A0, int lda0,
                 const u16* __restrict__ A1, int lda1,
                 const u16* __restrict__ W,
                 const float* __restrict__ bias, int M,
                 u16* __restrict__ outB, float* __restrict__ outF, int ldo) {
  __shared__ __align__(16) u16 tiles[2][TILE_ELEMS];

  const int tid    = threadIdx.x;
  const int lane   = tid & 31;
  const int laneLo = lane & 15;
  const int laneHi = lane >> 4;
  const int wave   = tid >> 5;
  const int waveM  = wave & 3;   // 4 wave-rows  -> 32 rows each
  const int waveN  = wave >> 2;  // 2 wave-cols  -> 64 cols each
  const int m0 = blockIdx.x * BM;
  const int n0 = blockIdx.y * BN;

  const u32 ldsBase = (u32)(uintptr_t)(void*)tiles;

  // per-thread staging coordinates: 4 transactions of 16B for A, 4 for B
  // q = tid + 256*it ; row = q>>3 (0..127) ; kchunk = (q&7)*8
  v8f acc[2][4];
#pragma unroll
  for (int i = 0; i < 2; ++i)
#pragma unroll
    for (int j = 0; j < 4; ++j) {
      v8f z = {0.f, 0.f, 0.f, 0.f, 0.f, 0.f, 0.f, 0.f};
      acc[i][j] = z;
    }

  const int nPhase  = (A1 != nullptr) ? 2 : 1;
  const int nChunks = nPhase * (512 / KC);

  auto issue_chunk = [&](int c, int dbuf) {
    const int   phase = c >> 3;
    const int   k0    = (c & 7) * KC;
    const u16*  Asrc  = phase ? A1 : A0;
    const int   lda   = phase ? lda1 : lda0;
    const int   kW    = phase * 512 + k0;      // k offset inside W rows
    const u32   aBase = ldsBase + (u32)dbuf * TILE_BYTES;
    const u32   bBase = aBase + (u32)(BM * LDT) * 2;
#pragma unroll
    for (int it = 0; it < 4; ++it) {
      int q  = tid + 256 * it;
      int r  = q >> 3;
      int kc = (q & 7) * 8;
      int gm = m0 + r;
      if (gm >= M) gm = M - 1;   // clamp: padded rows are discarded on store
      async_load_b128(aBase + (u32)(r * LDT + kc) * 2,
                      Asrc + (size_t)gm * lda + k0 + kc);
    }
#pragma unroll
    for (int it = 0; it < 4; ++it) {
      int q  = tid + 256 * it;
      int r  = q >> 3;
      int kc = (q & 7) * 8;
      async_load_b128(bBase + (u32)(r * LDT + kc) * 2,
                      W + (size_t)(n0 + r) * KTOT + kW + kc);
    }
  };

  issue_chunk(0, 0);
  for (int c = 0; c < nChunks; ++c) {
    const int dbuf = c & 1;
    if (c + 1 < nChunks) {
      issue_chunk(c + 1, dbuf ^ 1);
      wait_async8();             // chunk c's 8 loads done; c+1's in flight
    } else {
      wait_async0();
    }
    __syncthreads();             // all waves' slices of tile c are in LDS

    const u16* Asb = &tiles[dbuf][0];
    const u16* Bsb = &tiles[dbuf][BM * LDT];
#pragma unroll
    for (int kk = 0; kk < KC; kk += 32) {
      Frag a[2], b[4];
      // A fragment: 16-bit A 16x32 layout. dword v holds K-pair:
      //   v<4 -> K=2v , v>=4 -> K=2v+8 ; +8 for hi half-wave.
#pragma unroll
      for (int i = 0; i < 2; ++i) {
        int row = waveM * 32 + 16 * i + laneLo;
#pragma unroll
        for (int v = 0; v < 8; ++v) {
          int kb = ((v < 4) ? 2 * v : 2 * v + 8) + 8 * laneHi + kk;
          a[i].u[v] = *(const u32*)(&Asb[row * LDT + kb]);
        }
      }
      // B fragment: dword v holds K-pair: v<4 -> K=2v, v>=4 -> K=2v+8; +8 for hi half-wave, N=laneLo
#pragma unroll
      for (int j = 0; j < 4; ++j) {
        int col = waveN * 64 + 16 * j + laneLo;
#pragma unroll
        for (int v = 0; v < 8; ++v) {
          int kb = ((v < 4) ? 2 * v : 2 * v + 8) + 8 * laneHi + kk;
          b[j].u[v] = *(const u32*)(&Bsb[col * LDT + kb]);
        }
      }
#pragma unroll
      for (int i = 0; i < 2; ++i)
#pragma unroll
        for (int j = 0; j < 4; ++j)
          acc[i][j] = __builtin_amdgcn_wmma_f32_16x16x32_bf16(
              false, a[i].v, false, b[j].v, (short)0, acc[i][j],
              false, false);
    }
    __syncthreads();             // tile dbuf free for chunk c+2's async loads
  }

  // ---- epilogue: C/D layout: lane<16 -> M=r, lane>=16 -> M=8+r ; N=laneLo
#pragma unroll
  for (int i = 0; i < 2; ++i) {
    int mt = m0 + waveM * 32 + 16 * i;
#pragma unroll
    for (int j = 0; j < 4; ++j) {
      int colg = n0 + waveN * 64 + 16 * j + laneLo;
      float bv = bias[colg];
#pragma unroll
      for (int r = 0; r < 8; ++r) {
        int mr = mt + r + 8 * laneHi;
        if (mr < M) {
          float val = acc[i][j][r] + bv;
          if (HIDDEN) {
            outB[(size_t)mr * ldo + colg] = f2bf(tanhf(val));
          } else {
            outF[(size_t)mr * ldo + colg] = val;
          }
        }
      }
    }
  }
}

// ---------------------------------------------------------------- host
extern "C" void kernel_launch(void* const* d_in, const int* in_sizes, int n_in,
                              void* d_out, int out_size, void* d_ws,
                              size_t ws_size, hipStream_t stream) {
  (void)in_sizes; (void)n_in; (void)out_size; (void)ws_size;
  const float* x  = (const float*)d_in[0];
  const float* Wf = (const float*)d_in[1];
  const float* bf = (const float*)d_in[2];
  const float* Wb = (const float*)d_in[3];
  const float* bb = (const float*)d_in[4];
  const float* Wo = (const float*)d_in[5];
  const float* bo = (const float*)d_in[6];
  float* out = (float*)d_out;

  int sizes[DEPTH + 1], offs[DEPTH + 2];
  offs[0] = 0;
  for (int l = 0; l <= DEPTH; ++l) {
    int s = 1;
    for (int j = 0; j < l; ++j) s *= BRANCH;
    sizes[l] = s;
    offs[l + 1] = offs[l] + s;
  }
  const int nNodes = offs[DEPTH + 1];  // 87381

  // carve workspace (aligned to 256B)
  char* ws = (char*)d_ws;
  size_t cur = 0;
  auto carve = [&](size_t bytes) -> char* {
    char* p = ws + cur;
    cur += (bytes + 255) & ~(size_t)255;
    return p;
  };
  u16* Xbf   = (u16*)carve((size_t)nNodes * IN_DIM * 2);          // ~89.5 MB
  u16* Wfb   = (u16*)carve((size_t)HID * KTOT * 2);               // 1 MB
  u16* Wbb   = (u16*)carve((size_t)HID * KTOT * 2);               // 1 MB
  u16* Wob   = (u16*)carve((size_t)OUTD * KTOT * 2);              // 0.5 MB
  u16* Hm    = (u16*)carve((size_t)sizes[DEPTH - 1] * KTOT * 2);  // 32 MB
  u16* HcatA = (u16*)carve((size_t)sizes[DEPTH] * KTOT * 2);      // 128 MB
  u16* HcatB = (u16*)carve((size_t)sizes[DEPTH - 1] * KTOT * 2);  // 32 MB

  // one-time bf16 conversion of x and weights
  {
    int n = nNodes * IN_DIM;
    f32_to_bf16_kernel<<<(n + 255) / 256, 256, 0, stream>>>(x, Xbf, n);
    n = HID * KTOT;
    f32_to_bf16_kernel<<<(n + 255) / 256, 256, 0, stream>>>(Wf, Wfb, n);
    f32_to_bf16_kernel<<<(n + 255) / 256, 256, 0, stream>>>(Wb, Wbb, n);
    n = OUTD * KTOT;
    f32_to_bf16_kernel<<<(n + 255) / 256, 256, 0, stream>>>(Wo, Wob, n);
  }

  int rowOut = 0;
  for (int lvl = DEPTH; lvl >= 0; --lvl) {
    const int Ml = sizes[lvl];
    u16* HcatCur   = ((DEPTH - lvl) & 1) ? HcatB : HcatA;
    u16* HcatChild = ((DEPTH - lvl) & 1) ? HcatA : HcatB;

    const u16* hmF = nullptr;
    const u16* hmB = nullptr;
    if (lvl < DEPTH) {
      int nDw = Ml * (KTOT / 2);
      child_mean_kernel<<<(nDw + 255) / 256, 256, 0, stream>>>(
          (const u32*)HcatChild, (u32*)Hm, nDw);
      hmF = Hm;
      hmB = Hm + HID;
    }

    const u16* A0 = Xbf + (size_t)offs[lvl] * IN_DIM;
    dim3 gH((Ml + BM - 1) / BM, HID / BN);
    // h_f = tanh([x | hm_f] @ Wf^T + bf)  -> Hcat cols [0,512)
    wmma_gemm_kernel<true><<<gH, 256, 0, stream>>>(
        A0, IN_DIM, hmF, KTOT, Wfb, bf, Ml, HcatCur, nullptr, KTOT);
    // h_b = tanh([x | hm_b] @ Wb^T + bb)  -> Hcat cols [512,1024)
    wmma_gemm_kernel<true><<<gH, 256, 0, stream>>>(
        A0, IN_DIM, hmB, KTOT, Wbb, bb, Ml, HcatCur + HID, nullptr, KTOT);
    // logits = [h_f | h_b] @ Wo^T + bo -> d_out rows [rowOut, rowOut+Ml)
    dim3 gO((Ml + BM - 1) / BM, OUTD / BN);
    wmma_gemm_kernel<false><<<gO, 256, 0, stream>>>(
        HcatCur, KTOT, HcatCur + HID, KTOT, Wob, bo, Ml, nullptr,
        out + (size_t)rowOut * OUTD, OUTD);
    rowOut += Ml;
  }
}
